// MMDyConv2d_65764539236616
// MI455X (gfx1250) — matrix operand order repaired and checked by reading, not compiled, not run on hardware
//
#include <hip/hip_runtime.h>
#include <hip/hip_bf16.h>

// ---------------- common ----------------
#define B_   32
#define CIN  64
#define COUT 64
#define HH   128
#define WW   128
#define KK_  4
#define A_   16
#define TAPS 9
#define KRED (CIN * TAPS)   // 576

typedef __attribute__((ext_vector_type(16))) __bf16 v16bf;
typedef __attribute__((ext_vector_type(8)))  float  v8f;

__device__ inline unsigned short f2bf(float f) {
  unsigned int u = __float_as_uint(f);
  unsigned int r = u + 0x7FFFu + ((u >> 16) & 1u);   // round-to-nearest-even
  return (unsigned short)(r >> 16);
}
__device__ inline float sigmoidf(float x) { return 1.0f / (1.0f + __expf(-x)); }

// ---------------- kernel 1: global average pool ----------------
__global__ __launch_bounds__(256) void pool_kernel(const float* __restrict__ x,
                                                   float* __restrict__ pooled) {
  int bc = blockIdx.x;                       // b*CIN + ci
  const float* p = x + (size_t)bc * (HH * WW);
  float s = 0.0f;
  for (int i = threadIdx.x; i < HH * WW; i += 256) s += p[i];
  __shared__ float red[256];
  red[threadIdx.x] = s;
  __syncthreads();
  for (int off = 128; off > 0; off >>= 1) {
    if (threadIdx.x < off) red[threadIdx.x] += red[threadIdx.x + off];
    __syncthreads();
  }
  if (threadIdx.x == 0) pooled[bc] = red[0] * (1.0f / (HH * WW));
}

// ---------------- kernel 2: attention heads + fusion (1 block per sample) ----
__global__ __launch_bounds__(128) void attn_kernel(
    const float* __restrict__ pooled,
    const float* __restrict__ fc_w,  const float* __restrict__ ch_w,
    const float* __restrict__ ch_b,  const float* __restrict__ f_w,
    const float* __restrict__ f_b,   const float* __restrict__ sp_w,
    const float* __restrict__ sp_b,  const float* __restrict__ k_w,
    const float* __restrict__ k_b,   const float* __restrict__ fus_w1,
    const float* __restrict__ fus_b1,const float* __restrict__ fus_w2,
    const float* __restrict__ fus_b2,
    float* __restrict__ colScale,    // [B,CIN]  = w1 * ch_att
    float* __restrict__ rowScale,    // [B,COUT] = w2 * f_att
    float* __restrict__ spatt,       // [B,9]
    float* __restrict__ katt) {      // [B,4]
  __shared__ float pooled_s[CIN], a_s[A_], att_s[CIN + COUT], hid_s[8],
                   att2_s[CIN + COUT], klog_s[KK_], w_s[2];
  int b = blockIdx.x, tid = threadIdx.x;
  if (tid < CIN) pooled_s[tid] = pooled[b * CIN + tid];
  __syncthreads();
  if (tid < A_) {
    float s = 0.0f;
    for (int i = 0; i < CIN; ++i) s += pooled_s[i] * fc_w[tid * CIN + i];
    a_s[tid] = fmaxf(s, 0.0f);
  }
  __syncthreads();
  if (tid < CIN) {
    float s = ch_b[tid];
    for (int j = 0; j < A_; ++j) s += a_s[j] * ch_w[tid * A_ + j];
    att_s[tid] = sigmoidf(s);
  } else {
    int co = tid - CIN;
    float s = f_b[co];
    for (int j = 0; j < A_; ++j) s += a_s[j] * f_w[co * A_ + j];
    att_s[tid] = sigmoidf(s);
  }
  if (tid < TAPS) {
    float s = sp_b[tid];
    for (int j = 0; j < A_; ++j) s += a_s[j] * sp_w[tid * A_ + j];
    spatt[b * TAPS + tid] = sigmoidf(s);
  }
  if (tid < KK_) {
    float s = k_b[tid];
    for (int j = 0; j < A_; ++j) s += a_s[j] * k_w[tid * A_ + j];
    klog_s[tid] = s;
  }
  __syncthreads();
  if (tid == 0) {
    float m = klog_s[0];
    for (int k = 1; k < KK_; ++k) m = fmaxf(m, klog_s[k]);
    float den = 0.0f, e[KK_];
    for (int k = 0; k < KK_; ++k) { e[k] = __expf(klog_s[k] - m); den += e[k]; }
    for (int k = 0; k < KK_; ++k) katt[b * KK_ + k] = e[k] / den;
  }
  if (tid < 8) {
    float s = fus_b1[tid];
    for (int c = 0; c < CIN + COUT; ++c) s += att_s[c] * fus_w1[tid * (CIN + COUT) + c];
    hid_s[tid] = fmaxf(s, 0.0f);
  }
  __syncthreads();
  {
    float s = fus_b2[tid];
    for (int r = 0; r < 8; ++r) s += hid_s[r] * fus_w2[tid * 8 + r];
    att2_s[tid] = sigmoidf(s);
  }
  __syncthreads();
  if (tid == 0) {
    float s1 = 0.0f, s2 = 0.0f;
    for (int i = 0; i < CIN; ++i)  s1 += att2_s[i];
    for (int i = CIN; i < CIN + COUT; ++i) s2 += att2_s[i];
    float m = fmaxf(s1, s2);
    float e1 = __expf(s1 - m), e2 = __expf(s2 - m);
    w_s[0] = e1 / (e1 + e2);
    w_s[1] = e2 / (e1 + e2);
  }
  __syncthreads();
  if (tid < CIN) colScale[b * CIN + tid] = w_s[0] * att_s[tid];
  else           rowScale[b * COUT + (tid - CIN)] = w_s[1] * att_s[tid];
}

// ---------------- kernel 3: fold everything into per-sample bf16 weights -----
// wbf16[b][co][t*64 + ci] (K-dim ordered tap-major so the conv loops taps outer)
__global__ __launch_bounds__(256) void wagg_kernel(
    const float* __restrict__ weight,   // (K, COUT, CIN, 3, 3)
    const float* __restrict__ colScale, const float* __restrict__ rowScale,
    const float* __restrict__ spatt,    const float* __restrict__ katt,
    unsigned short* __restrict__ wbf16) {
  int idx = blockIdx.x * 256 + threadIdx.x;            // B*COUT*CIN*9 total
  int b  = idx / (COUT * CIN * TAPS);
  int r  = idx - b * (COUT * CIN * TAPS);
  int co = r / (CIN * TAPS);
  int r2 = r - co * (CIN * TAPS);
  int ci = r2 / TAPS;
  int t  = r2 - ci * TAPS;
  float acc = 0.0f;
  #pragma unroll
  for (int k = 0; k < KK_; ++k)
    acc += katt[b * KK_ + k] * weight[(((size_t)k * COUT + co) * CIN + ci) * TAPS + t];
  float v = acc * rowScale[b * COUT + co] * colScale[b * CIN + ci] * spatt[b * TAPS + t];
  wbf16[((size_t)b * COUT + co) * KRED + t * CIN + ci] = f2bf(v);
}

// ---------------- kernel 4: per-sample implicit-GEMM conv via bf16 WMMA ------
// Grid: B*H blocks; each block = 8 waves computes out[b, 0:64co, h, 0:128w].
#define XCOLS 130           // w = -1 .. 128 (halo)
#define CIPAD 66            // ci stride padded (odd dword stride -> no bank conflicts)
#define APAD  33            // dwords per co row of A tile
__global__ __launch_bounds__(256) void conv_wmma_kernel(
    const float* __restrict__ x, const unsigned short* __restrict__ wagg,
    float* __restrict__ out) {
  __shared__ unsigned short xs[3 * XCOLS * CIPAD];   // [(r*130+col)*66 + ci], bf16 bits
  __shared__ unsigned int   as[COUT * APAD];         // [co*33 + ci/2], one tap

  int b = blockIdx.x >> 7;
  int h = blockIdx.x & (HH - 1);
  int tid = threadIdx.x;

  // ---- stage 3 input rows (all 64 ci) as bf16, zero-padded halo ----
  const int NX = CIN * 3 * XCOLS;
  for (int idx = tid; idx < NX; idx += 256) {
    int ci  = idx / (3 * XCOLS);
    int rem = idx - ci * (3 * XCOLS);
    int r   = rem / XCOLS;
    int c   = rem - r * XCOLS;
    int gh = h + r - 1;
    int gw = c - 1;
    float v = 0.0f;
    if ((unsigned)gh < (unsigned)HH && (unsigned)gw < (unsigned)WW)
      v = x[(((size_t)b * CIN + ci) * HH + gh) * WW + gw];
    xs[(r * XCOLS + c) * CIPAD + ci] = f2bf(v);
  }

  int wave = tid >> 5;
  int lane = tid & 31;
  int l16  = lane & 15;
  int hi   = lane >> 4;           // 0: lanes 0-15, 1: lanes 16-31
  int m_tile = wave & 3;          // co block (16 rows)
  int n_grp  = wave >> 2;         // pixel half: 0 -> w 0..63, 1 -> w 64..127

  union BF { v16bf v; unsigned int u[8]; };
  v8f acc[4] = {};                // 4 n-tiles of 16 pixels each

  const unsigned int* wb = (const unsigned int*)(wagg + (size_t)b * COUT * KRED);

  for (int t = 0; t < TAPS; ++t) {
    int dy = t / 3, dx = t % 3;   // input row r = dy, local col = w + dx
    __syncthreads();              // protect `as` reuse (also covers xs on t==0)
    // load A tap tile: 64co x 64ci bf16 = 2048 dwords
    for (int i = tid; i < COUT * 32; i += 256) {
      int co = i >> 5;
      int p  = i & 31;
      as[co * APAD + p] = wb[co * (KRED / 2) + t * 32 + p];
    }
    __syncthreads();

    #pragma unroll
    for (int cc = 0; cc < 2; ++cc) {          // ci chunks of 32
      int ci0 = cc * 32;
      // A fragment: 16co x 32k, contiguous k order (lane<16: k 0-15; lane>=16: k 16-31)
      BF afrag;
      int m  = m_tile * 16 + l16;
      int kb = hi * 16;
      #pragma unroll
      for (int v = 0; v < 8; ++v) {
        int k = kb + 2 * v;
        afrag.u[v] = as[m * APAD + ((ci0 + k) >> 1)];
      }
      #pragma unroll
      for (int j = 0; j < 4; ++j) {           // this wave's 4 pixel tiles
        // B fragment: 32k x 16px, contiguous k order (lane<16: k 0-15; lane>=16: k 16-31)
        BF bfrag;
        int n    = n_grp * 64 + j * 16 + l16; // output w
        int base = (dy * XCOLS + (n + dx)) * (CIPAD / 2);   // dword index
        int kb2  = hi * 16;
        #pragma unroll
        for (int v = 0; v < 8; ++v)
          bfrag.u[v] = ((const unsigned int*)xs)[base + ((ci0 + kb2) >> 1) + v];
        acc[j] = __builtin_amdgcn_wmma_f32_16x16x32_bf16(
            false, afrag.v, false, bfrag.v, (short)0, acc[j], false, false);
      }
    }
  }

  // ---- writeback: C/D layout — VGPR r: lanes 0-15 M=r, lanes 16-31 M=8+r ----
  #pragma unroll
  for (int j = 0; j < 4; ++j) {
    int wpx = n_grp * 64 + j * 16 + l16;
    #pragma unroll
    for (int r = 0; r < 8; ++r) {
      int co = m_tile * 16 + hi * 8 + r;
      out[(((size_t)b * COUT + co) * HH + h) * WW + wpx] = acc[j][r];
    }
  }
}

// ---------------- launcher ----------------
extern "C" void kernel_launch(void* const* d_in, const int* in_sizes, int n_in,
                              void* d_out, int out_size, void* d_ws, size_t ws_size,
                              hipStream_t stream) {
  const float* x      = (const float*)d_in[0];
  const float* fc_w   = (const float*)d_in[1];
  const float* ch_w   = (const float*)d_in[2];
  const float* ch_b   = (const float*)d_in[3];
  const float* f_w    = (const float*)d_in[4];
  const float* f_b    = (const float*)d_in[5];
  const float* sp_w   = (const float*)d_in[6];
  const float* sp_b   = (const float*)d_in[7];
  const float* k_w    = (const float*)d_in[8];
  const float* k_b    = (const float*)d_in[9];
  const float* fus_w1 = (const float*)d_in[10];
  const float* fus_b1 = (const float*)d_in[11];
  const float* fus_w2 = (const float*)d_in[12];
  const float* fus_b2 = (const float*)d_in[13];
  const float* weight = (const float*)d_in[14];
  float* out = (float*)d_out;

  char* ws = (char*)d_ws;
  float* pooled   = (float*)(ws + 0);          // 2048 f
  float* colScale = (float*)(ws + 8192);       // 2048 f
  float* rowScale = (float*)(ws + 16384);      // 2048 f
  float* spatt    = (float*)(ws + 24576);      // 288 f
  float* katt     = (float*)(ws + 25728);      // 128 f
  unsigned short* wbf16 = (unsigned short*)(ws + 32768);  // B*64*576 bf16

  pool_kernel<<<B_ * CIN, 256, 0, stream>>>(x, pooled);
  attn_kernel<<<B_, 128, 0, stream>>>(pooled, fc_w, ch_w, ch_b, f_w, f_b,
                                      sp_w, sp_b, k_w, k_b, fus_w1, fus_b1,
                                      fus_w2, fus_b2, colScale, rowScale,
                                      spatt, katt);
  wagg_kernel<<<(B_ * COUT * CIN * TAPS) / 256, 256, 0, stream>>>(
      weight, colScale, rowScale, spatt, katt, wbf16);
  conv_wmma_kernel<<<B_ * HH, 256, 0, stream>>>(x, wbf16, out);
}
